// JTMPN_13417477833135
// MI455X (gfx1250) — compile-verified, matrix operand order checked by reading, not run on hardware
//
#include <hip/hip_runtime.h>
#include <hip/hip_bf16.h>

#define ATOM_FDIM 35
#define BOND_IN   40      // ATOM_FDIM + BOND_FDIM
#define MAX_NB    10
#define HIDDEN    128
#define DEPTH     6
#define N_ATOMS   100000
#define N_BONDS   220000
#define N_MESS    20000
#define N_MOLS    2000
#define MSG_ROWS  (N_MESS + N_BONDS)
#define K_ATOM    192     // 163 padded to 6*32

typedef __attribute__((ext_vector_type(16))) __bf16 v16bf;
typedef __attribute__((ext_vector_type(8)))  float  v8f;
typedef __attribute__((ext_vector_type(2)))  __bf16 v2bf;

struct bfrag_bits { uint4 lo, hi; };   // 32 bytes == one v16bf

__device__ __forceinline__ unsigned short f2bf(float f) {
  unsigned u = __builtin_bit_cast(unsigned, f);
  unsigned r = u + 0x7FFFu + ((u >> 16) & 1u);   // round-to-nearest-even
  return (unsigned short)(r >> 16);
}
__device__ __forceinline__ float bf_lo(unsigned q) {   // low bf16 of a dword
  return __builtin_bit_cast(float, q << 16);
}
__device__ __forceinline__ float bf_hi(unsigned q) {   // high bf16 of a dword
  return __builtin_bit_cast(float, q & 0xFFFF0000u);
}
__device__ __forceinline__ unsigned pack2(float a, float b) {
#if __has_builtin(__builtin_amdgcn_cvt_pk_bf16_f32)
  v2bf p = __builtin_amdgcn_cvt_pk_bf16_f32(a, b);   // single v_cvt_pk_bf16_f32
  return __builtin_bit_cast(unsigned, p);
#else
  return (unsigned)f2bf(a) | ((unsigned)f2bf(b) << 16);
#endif
}

// A-matrix 16xK fragment from an LDS tile (row-major bf16, stride in elems).
// Per-lane data = two contiguous 8-elem runs -> two ds_load_b128.
__device__ __forceinline__ v16bf frag_a(const unsigned short* lds, int stride,
                                        int lane, int kb) {
  const int m = lane & 15, half = lane >> 4;
  const unsigned short* p = lds + m * stride + kb + 8 * half;
  bfrag_bits pr;
  pr.lo = *(const uint4*)p;          // K = kb+8h .. +7
  pr.hi = *(const uint4*)(p + 16);   // K = kb+16+8h .. +7
  return __builtin_bit_cast(v16bf, pr);
}

// B fragment from pre-swizzled global weights: [(nb*KT+kt)][lane][16 bf16].
__device__ __forceinline__ v16bf load_bfrag(const unsigned short* __restrict__ g,
                                            int nb, int KT, int kt, int lane) {
  const unsigned short* p = g + (size_t)(((nb * KT + kt) * 32 + lane)) * 16;
  bfrag_bits pr = *(const bfrag_bits*)p;   // 32B, coalesced across the wave
  return __builtin_bit_cast(v16bf, pr);
}

// ---------------- small prep kernels ----------------

__global__ void zero_kernel(float* out, float* counts) {
  int i = blockIdx.x * blockDim.x + threadIdx.x;
  if (i < N_MOLS * HIDDEN) out[i] = 0.f;
  if (i < N_MOLS) counts[i] = 0.f;
}

// Swizzle one weight matrix into per-lane WMMA B-fragment order.
__device__ __forceinline__ void swizzle_one(const float* __restrict__ W,
                                            unsigned short* __restrict__ out,
                                            int i, int KT, int realK, bool atomPerm) {
  int e    = i & 15;
  int lane = (i >> 4) & 31;
  int f    = i >> 9;
  int kt = f % KT, nb = f / KT;
  int half = lane >> 4;
  int koff = (e < 8) ? (e + 8 * half) : (e + 8 + 8 * half);
  int k = kt * 32 + koff;
  int n = nb * 16 + (lane & 15);
  if (atomPerm) {
    // atom K order: [nei(128) -> W rows 35..162 | fatoms(35) -> W rows 0..34 | pad]
    k = (k < HIDDEN) ? (ATOM_FDIM + k) : (k < 163 ? (k - HIDDEN) : realK);
  }
  out[i] = (k < realK) ? f2bf(W[k * HIDDEN + n]) : (unsigned short)0;
}

__global__ void prep_kernel(const float* __restrict__ Wi, const float* __restrict__ Wh,
                            const float* __restrict__ Wo,
                            unsigned short* WiS, unsigned short* WhS, unsigned short* WoS) {
  int i = blockIdx.x * blockDim.x + threadIdx.x;
  if (i < 8 * 2 * 32 * 16) swizzle_one(Wi, WiS, i, 2, BOND_IN, false);
  if (i < 8 * 4 * 32 * 16) swizzle_one(Wh, WhS, i, 4, HIDDEN, false);
  if (i < 8 * 6 * 32 * 16) swizzle_one(Wo, WoS, i, 6, 163, true);
}

__global__ void tree_kernel(const float* __restrict__ tree,
                            unsigned short* msgA, unsigned short* msgB) {
  int i = blockIdx.x * blockDim.x + threadIdx.x;
  if (i < N_MESS * HIDDEN) { unsigned short h = f2bf(tree[i]); msgA[i] = h; msgB[i] = h; }
}

__global__ void count_kernel(const int* __restrict__ mol_ids, float* counts) {
  int i = blockIdx.x * blockDim.x + threadIdx.x;
  if (i < N_ATOMS) atomicAdd(&counts[mol_ids[i]], 1.0f);
}

__global__ void fin_kernel(float* out, const float* __restrict__ counts) {
  int i = blockIdx.x * blockDim.x + threadIdx.x;
  if (i < N_MOLS * HIDDEN) {
    float c = counts[i >> 7];
    out[i] /= (c > 1.0f ? c : 1.0f);
  }
}

// ---------------- binput = fbonds @ W_i ----------------

__global__ __launch_bounds__(128)
void binput_kernel(const float* __restrict__ fbonds, const unsigned short* __restrict__ WiS,
                   float* __restrict__ binput, unsigned short* __restrict__ msgA) {
  __shared__ __align__(16) unsigned short fb[16 * 64];     //  2 KB
  __shared__ __align__(16) float          op[16 * HIDDEN]; //  8 KB
  const int tid = threadIdx.x, lane = tid & 31, wave = tid >> 5;
  const int c4 = (tid & 31) * 4, rg = tid >> 5;

  v16bf bh[2][2];
#pragma unroll
  for (int nt = 0; nt < 2; ++nt)
#pragma unroll
    for (int kt = 0; kt < 2; ++kt)
      bh[nt][kt] = load_bfrag(WiS, wave * 2 + nt, 2, kt, lane);

  const int nTiles = N_BONDS / 16;
  for (int t = blockIdx.x; t < nTiles; t += gridDim.x) {
    const int base = t * 16;
    for (int i = tid; i < 16 * 64; i += 128) {
      int r = i >> 6, c = i & 63;
      fb[i] = (c < BOND_IN) ? f2bf(fbonds[(size_t)(base + r) * BOND_IN + c]) : (unsigned short)0;
    }
    __syncthreads();
    v16bf a0 = frag_a(fb, 64, lane, 0);
    v16bf a1 = frag_a(fb, 64, lane, 32);
#pragma unroll
    for (int nt = 0; nt < 2; ++nt) {
      const int nb = (wave * 2 + nt) * 16;
      v8f acc = {};
      acc = __builtin_amdgcn_wmma_f32_16x16x32_bf16(false, a0, false, bh[nt][0], (short)0, acc, false, false);
      acc = __builtin_amdgcn_wmma_f32_16x16x32_bf16(false, a1, false, bh[nt][1], (short)0, acc, false, false);
      const int n = nb + (lane & 15), half = lane >> 4;
#pragma unroll
      for (int r = 0; r < 8; ++r) op[(r + 8 * half) * HIDDEN + n] = acc[r];
    }
    __syncthreads();
#pragma unroll
    for (int i = 0; i < 4; ++i) {
      int r = rg + 4 * i;
      float4 v = *(const float4*)(op + r * HIDDEN + c4);
      *(float4*)(binput + (size_t)(base + r) * HIDDEN + c4) = v;
      uint2 o;
      o.x = pack2(fmaxf(v.x, 0.f), fmaxf(v.y, 0.f));
      o.y = pack2(fmaxf(v.z, 0.f), fmaxf(v.w, 0.f));
      *(uint2*)(msgA + (size_t)(N_MESS + base + r) * HIDDEN + c4) = o;
    }
    __syncthreads();
  }
}

// -------- one depth step: relu(binput + gathersum(msg)[bgraph] @ W_h) --------

__global__ __launch_bounds__(128)
void mp_kernel(const unsigned short* __restrict__ msgSrc, unsigned short* __restrict__ msgDst,
               const float* __restrict__ binput, const int* __restrict__ bgraph,
               const unsigned short* __restrict__ WhS) {
  __shared__ __align__(16) unsigned short nei[16 * HIDDEN];  // 4 KB
  __shared__ __align__(16) float          bin[16 * HIDDEN];  // 8 KB
  __shared__ __align__(16) unsigned short op[16 * HIDDEN];   // 4 KB
  __shared__ __align__(16) int            idx[16 * MAX_NB];
  const int tid = threadIdx.x, lane = tid & 31, wave = tid >> 5;
  const int c4 = (tid & 31) * 4, rg = tid >> 5;

  v16bf bh[2][4];                       // W_h fragments live in registers
#pragma unroll
  for (int nt = 0; nt < 2; ++nt)
#pragma unroll
    for (int kt = 0; kt < 4; ++kt)
      bh[nt][kt] = load_bfrag(WhS, wave * 2 + nt, 4, kt, lane);

  const int nTiles = N_BONDS / 16;
  for (int t = blockIdx.x; t < nTiles; t += gridDim.x) {
    const int base = t * 16;
    if (tid < 40) *(int4*)(idx + tid * 4) = *(const int4*)(bgraph + (size_t)base * MAX_NB + tid * 4);
#pragma unroll
    for (int i = 0; i < 4; ++i) {
      int r = rg + 4 * i;
      *(float4*)(bin + r * HIDDEN + c4) = *(const float4*)(binput + (size_t)(base + r) * HIDDEN + c4);
    }
    __syncthreads();
#pragma unroll
    for (int i = 0; i < 4; ++i) {       // gather-sum: 4 cols x 4 rows per thread
      int r = rg + 4 * i;
      float s0 = 0.f, s1 = 0.f, s2 = 0.f, s3 = 0.f;
#pragma unroll
      for (int j = 0; j < MAX_NB; ++j) {
        uint2 q = *(const uint2*)(msgSrc + (size_t)idx[r * MAX_NB + j] * HIDDEN + c4);
        s0 += bf_lo(q.x); s1 += bf_hi(q.x);
        s2 += bf_lo(q.y); s3 += bf_hi(q.y);
      }
      uint2 o; o.x = pack2(s0, s1); o.y = pack2(s2, s3);
      *(uint2*)(nei + r * HIDDEN + c4) = o;
    }
    __syncthreads();
    v16bf afr[4];
#pragma unroll
    for (int kt = 0; kt < 4; ++kt) afr[kt] = frag_a(nei, HIDDEN, lane, kt * 32);
#pragma unroll
    for (int nt = 0; nt < 2; ++nt) {
      const int nb = (wave * 2 + nt) * 16;
      v8f acc = {};
#pragma unroll
      for (int kt = 0; kt < 4; ++kt)
        acc = __builtin_amdgcn_wmma_f32_16x16x32_bf16(false, afr[kt], false, bh[nt][kt],
                                                      (short)0, acc, false, false);
      const int n = nb + (lane & 15), half = lane >> 4;
#pragma unroll
      for (int r = 0; r < 8; r += 2) {
        int row0 = r + 8 * half, row1 = row0 + 1;
        float v0 = fmaxf(acc[r]     + bin[row0 * HIDDEN + n], 0.f);
        float v1 = fmaxf(acc[r + 1] + bin[row1 * HIDDEN + n], 0.f);
        op[row0 * HIDDEN + n] = f2bf(v0);
        op[row1 * HIDDEN + n] = f2bf(v1);
      }
    }
    __syncthreads();
#pragma unroll
    for (int i = 0; i < 4; ++i) {
      int r = rg + 4 * i;
      *(uint2*)(msgDst + (size_t)(N_MESS + base + r) * HIDDEN + c4) =
          *(const uint2*)(op + r * HIDDEN + c4);
    }
    __syncthreads();
  }
}

// ------ atoms: relu([nei | fatoms] @ permuted W_o + b_o), fused segment-sum ------

__global__ __launch_bounds__(128)
void atom_kernel(const float* __restrict__ fatoms, const unsigned short* __restrict__ msg,
                 const int* __restrict__ agraph, const int* __restrict__ mol_ids,
                 const unsigned short* __restrict__ WoS, const float* __restrict__ b_o,
                 float* __restrict__ outSums) {
  __shared__ __align__(16) unsigned short ain[16 * K_ATOM];  // 6 KB
  __shared__ __align__(16) int idx[16 * MAX_NB];
  __shared__ int mol[16];
  const int tid = threadIdx.x, lane = tid & 31, wave = tid >> 5;
  const int c4 = (tid & 31) * 4, rg = tid >> 5;

  v16bf bh[2][6];
#pragma unroll
  for (int nt = 0; nt < 2; ++nt)
#pragma unroll
    for (int kt = 0; kt < 6; ++kt)
      bh[nt][kt] = load_bfrag(WoS, wave * 2 + nt, 6, kt, lane);

  const int nTiles = N_ATOMS / 16;
  for (int t = blockIdx.x; t < nTiles; t += gridDim.x) {
    const int base = t * 16;
    if (tid < 40) *(int4*)(idx + tid * 4) = *(const int4*)(agraph + (size_t)base * MAX_NB + tid * 4);
    if (tid < 16) mol[tid] = mol_ids[base + tid];
    // fatoms -> K cols 128..162 ; pad -> 163..191
    for (int i = tid; i < 16 * ATOM_FDIM; i += 128) {
      int r = i / ATOM_FDIM, c = i % ATOM_FDIM;
      ain[r * K_ATOM + HIDDEN + c] = f2bf(fatoms[(size_t)(base + r) * ATOM_FDIM + c]);
    }
    for (int i = tid; i < 16 * (K_ATOM - 163); i += 128) {
      int r = i / (K_ATOM - 163), c = i % (K_ATOM - 163);
      ain[r * K_ATOM + 163 + c] = 0;
    }
    __syncthreads();
#pragma unroll
    for (int i = 0; i < 4; ++i) {       // gathered hidden -> K cols 0..127
      int r = rg + 4 * i;
      float s0 = 0.f, s1 = 0.f, s2 = 0.f, s3 = 0.f;
#pragma unroll
      for (int j = 0; j < MAX_NB; ++j) {
        uint2 q = *(const uint2*)(msg + (size_t)idx[r * MAX_NB + j] * HIDDEN + c4);
        s0 += bf_lo(q.x); s1 += bf_hi(q.x);
        s2 += bf_lo(q.y); s3 += bf_hi(q.y);
      }
      uint2 o; o.x = pack2(s0, s1); o.y = pack2(s2, s3);
      *(uint2*)(ain + r * K_ATOM + c4) = o;
    }
    __syncthreads();
    v16bf afr[6];
#pragma unroll
    for (int kt = 0; kt < 6; ++kt) afr[kt] = frag_a(ain, K_ATOM, lane, kt * 32);
#pragma unroll
    for (int nt = 0; nt < 2; ++nt) {
      const int nb = (wave * 2 + nt) * 16;
      v8f acc = {};
#pragma unroll
      for (int kt = 0; kt < 6; ++kt)
        acc = __builtin_amdgcn_wmma_f32_16x16x32_bf16(false, afr[kt], false, bh[nt][kt],
                                                      (short)0, acc, false, false);
      const int n = nb + (lane & 15), half = lane >> 4;
      const float bias = b_o[n];
#pragma unroll
      for (int r = 0; r < 8; ++r) {
        int row = r + 8 * half;
        float v = fmaxf(acc[r] + bias, 0.f);
        atomicAdd(&outSums[(size_t)mol[row] * HIDDEN + n], v);
      }
    }
    __syncthreads();
  }
}

extern "C" void kernel_launch(void* const* d_in, const int* in_sizes, int n_in,
                              void* d_out, int out_size, void* d_ws, size_t ws_size,
                              hipStream_t stream) {
  (void)in_sizes; (void)n_in; (void)out_size; (void)ws_size;
  const float* fatoms = (const float*)d_in[0];
  const float* fbonds = (const float*)d_in[1];
  const float* tree   = (const float*)d_in[2];
  const int*   agraph = (const int*)d_in[3];
  const int*   bgraph = (const int*)d_in[4];
  const int*   molid  = (const int*)d_in[5];
  const float* Wi     = (const float*)d_in[7];
  const float* Wh     = (const float*)d_in[8];
  const float* Wo     = (const float*)d_in[9];
  const float* bo     = (const float*)d_in[10];
  float* out = (float*)d_out;

  char* ws = (char*)d_ws;
  size_t off = 0;
  auto take = [&](size_t bytes) { void* p = ws + off; off += (bytes + 255) & ~(size_t)255; return p; };
  unsigned short* msgA = (unsigned short*)take((size_t)MSG_ROWS * HIDDEN * 2); // 61.4 MB (L2-resident)
  unsigned short* msgB = (unsigned short*)take((size_t)MSG_ROWS * HIDDEN * 2); // 61.4 MB
  float*          binp = (float*)take((size_t)N_BONDS * HIDDEN * 4);           // 112.6 MB
  unsigned short* WiS  = (unsigned short*)take(8 * 2 * 32 * 16 * 2);
  unsigned short* WhS  = (unsigned short*)take(8 * 4 * 32 * 16 * 2);
  unsigned short* WoS  = (unsigned short*)take(8 * 6 * 32 * 16 * 2);
  float*          cnt  = (float*)take(N_MOLS * 4);

  zero_kernel <<<(N_MOLS * HIDDEN + 255) / 256, 256, 0, stream>>>(out, cnt);
  prep_kernel <<<(8 * 6 * 32 * 16 + 255) / 256, 256, 0, stream>>>(Wi, Wh, Wo, WiS, WhS, WoS);
  tree_kernel <<<(N_MESS * HIDDEN + 255) / 256, 256, 0, stream>>>(tree, msgA, msgB);
  count_kernel<<<(N_ATOMS + 255) / 256, 256, 0, stream>>>(molid, cnt);

  binput_kernel<<<2048, 128, 0, stream>>>(fbonds, WiS, binp, msgA);

  unsigned short *src = msgA, *dst = msgB;
  for (int d = 0; d < DEPTH - 1; ++d) {
    mp_kernel<<<2048, 128, 0, stream>>>(src, dst, binp, bgraph, WhS);
    unsigned short* tmp = src; src = dst; dst = tmp;
  }
  // after 5 steps the freshest message buffer is `src` (== msgB)
  atom_kernel<<<2048, 128, 0, stream>>>(fatoms, src, agraph, molid, WoS, bo, out);
  fin_kernel  <<<(N_MOLS * HIDDEN + 255) / 256, 256, 0, stream>>>(out, cnt);
}
